// TopKQueryBessKGE_14663018348644
// MI455X (gfx1250) — matrix operand order repaired and checked by reading, not compiled, not run on hardware
//
#include <hip/hip_runtime.h>
#include <hip/hip_bf16.h>

// ---------------- problem constants (from reference) ----------------
#define SDIM   4
#define BDIM   256
#define PDIM   32768
#define DDIM   256
#define QDIM   (SDIM * BDIM)       // 1024
#define KOUT   10
#define NBEST  11                  // K+1 running candidates per (s_ent, q)
#define BADV   (-10000.0f)

#define QT     16                  // queries per workgroup (one WMMA M tile)
#define NT     128                 // entities per strip (8 waves x 16 N)
#define NSTRIP (PDIM / NT)         // 256 strips
#define CANDW  (2 * NT)            // candidate queue per row covers 2 strips
#define NEGINF (-3.0e38f)

typedef __attribute__((ext_vector_type(16))) __bf16 v16bf;
typedef __attribute__((ext_vector_type(8)))  __bf16 v8bf;
typedef __attribute__((ext_vector_type(8)))  float  v8f;

// ---------------- kernel 1: fp32 -> bf16 entity table ----------------
__global__ __launch_bounds__(256)
void cvt_ent_kernel(const float* __restrict__ ent, __bf16* __restrict__ out) {
    size_t i = ((size_t)blockIdx.x * 256 + threadIdx.x) * 4;   // total = S*P*D
    float4 v = *(const float4*)(ent + i);
    out[i + 0] = (__bf16)v.x;
    out[i + 1] = (__bf16)v.y;
    out[i + 2] = (__bf16)v.z;
    out[i + 3] = (__bf16)v.w;
}

// ---------------- kernel 2: qvec = head_emb * rel_emb (bf16) ----------------
__global__ __launch_bounds__(256)
void build_qvec_kernel(const int* __restrict__ relation, const int* __restrict__ head,
                       const float* __restrict__ ent, const float* __restrict__ rel_emb,
                       __bf16* __restrict__ qvec) {
    int q = blockIdx.x;            // 0..1023 (= s_q*B + b, matches flattened (S,B))
    int d = threadIdx.x;           // 0..255
    int s = q >> 8;                // q / BDIM
    int h = head[q];
    int r = relation[q];
    float v = ent[((size_t)s * PDIM + h) * DDIM + d] * rel_emb[(size_t)r * DDIM + d];
    qvec[(size_t)q * DDIM + d] = (__bf16)v;
}

// ---------------- helpers for kernel 3 ----------------
__device__ __forceinline__
void load_bstrip(v16bf b[8], const __bf16* __restrict__ entbase,
                 int nb, int wave, int lane) {
    // B 32x16 bf16 fragment: lane l -> column n = l&15; K group (l>=16 ? 16..31 : 0..15)
    const int hi = lane >> 4;
    const __bf16* brow = entbase + (size_t)(nb + wave * 16 + (lane & 15)) * DDIM + hi * 16;
    #pragma unroll
    for (int ks = 0; ks < 8; ++ks)
        b[ks] = *(const v16bf*)(brow + ks * 32);   // 32 contiguous bytes (2 x b128)
}

__device__ __forceinline__
v8f gemm_tile(const v16bf a[8], const v16bf b[8]) {
    v8f c = {};
    #pragma unroll
    for (int ks = 0; ks < 8; ++ks)
        c = __builtin_amdgcn_wmma_f32_16x16x32_bf16(
                false, a[ks], false, b[ks], (short)0, c, false, false);
    return c;
}

// ---------------- kernel 3: fused GEMM (bf16 WMMA) + running top-11 ----------------
__global__ __launch_bounds__(256)
void gemm_topk_kernel(const __bf16* __restrict__ entbf, const __bf16* __restrict__ qvec,
                      float* __restrict__ best_sc, int* __restrict__ best_ix) {
    __shared__ float topsc[QT][NBEST];        // sorted descending per row
    __shared__ int   topix[QT][NBEST];
    __shared__ float candsc[QT][CANDW];       // survivors of up to 2 strips
    __shared__ int   candix[QT][CANDW];
    __shared__ int   candcnt[QT];

    const int tid   = threadIdx.x;
    const int wave  = tid >> 5;
    const int lane  = tid & 31;
    const int s_ent = blockIdx.x >> 6;        // 4 entity slices
    const int qbase = (blockIdx.x & 63) * QT;
    const int m     = lane & 15;
    const int hi    = lane >> 4;

    if (tid < QT) {
        candcnt[tid] = 0;
        #pragma unroll
        for (int j = 0; j < NBEST; ++j) { topsc[tid][j] = NEGINF; topix[tid][j] = PDIM; }
    }
    __syncthreads();

    // ---- preload A fragments for the full K=256, in contiguous k order, same as B ----
    // lane l: row M = l&15; K group {k0..k0+15}, k0 = 32*ks + 16*(l>>4)
    const __bf16* arow = qvec + (size_t)(qbase + m) * DDIM;
    v16bf afrag[8];
    #pragma unroll
    for (int ks = 0; ks < 8; ++ks) {
        int k0 = ks * 32 + hi * 16;
        v8bf lo = *(const v8bf*)(arow + k0);
        v8bf hh = *(const v8bf*)(arow + k0 + 8);
        afrag[ks] = __builtin_shufflevector(lo, hh, 0,1,2,3,4,5,6,7,8,9,10,11,12,13,14,15);
    }

    const __bf16* entbase = entbf + (size_t)s_ent * PDIM * DDIM;

    // ---- double-buffered B with 2-strip lookahead (register prefetch) ----
    v16bf b0[8], b1[8];
    load_bstrip(b0, entbase, 0,  wave, lane);
    load_bstrip(b1, entbase, NT, wave, lane);

    #pragma unroll 1
    for (int strip = 0; strip < NSTRIP; strip += 2) {
        // -------- half 0: GEMM + prefetch + register filter (no barrier) --------
        {
            const int nb = strip * NT;
            v8f c = gemm_tile(afrag, b0);
            int pre = strip + 2;
            load_bstrip(b0, entbase, (pre < NSTRIP ? pre : 0) * NT, wave, lane);

            const int ncol = nb + wave * 16 + (lane & 15);
            #pragma unroll
            for (int r = 0; r < 8; ++r) {
                int row = r + hi * 8;
                float s = c[r];
                if (s > topsc[row][NBEST - 1]) {
                    int pos = atomicAdd(&candcnt[row], 1);
                    candsc[row][pos] = s;
                    candix[row][pos] = ncol;
                }
            }
        }
        // -------- half 1: GEMM + prefetch + register filter --------
        {
            const int nb = (strip + 1) * NT;
            v8f c = gemm_tile(afrag, b1);
            int pre = strip + 3;
            load_bstrip(b1, entbase, (pre < NSTRIP ? pre : 0) * NT, wave, lane);

            const int ncol = nb + wave * 16 + (lane & 15);
            #pragma unroll
            for (int r = 0; r < 8; ++r) {
                int row = r + hi * 8;
                float s = c[r];
                if (s > topsc[row][NBEST - 1]) {
                    int pos = atomicAdd(&candcnt[row], 1);
                    candsc[row][pos] = s;
                    candix[row][pos] = ncol;
                }
            }
        }
        __syncthreads();

        // -------- merge survivors of both strips (rare after warm-up) --------
        if (tid < QT) {
            int row = tid;
            int nc  = candcnt[row];
            for (int j = 0; j < nc; ++j) {
                float s = candsc[row][j];
                if (s > topsc[row][NBEST - 1]) {
                    int p = NBEST - 1;
                    while (p > 0 && topsc[row][p - 1] < s) {
                        topsc[row][p] = topsc[row][p - 1];
                        topix[row][p] = topix[row][p - 1];
                        --p;
                    }
                    topsc[row][p] = s;
                    topix[row][p] = candix[row][j];
                }
            }
            candcnt[row] = 0;
        }
        __syncthreads();
    }

    // ---- emit per-(s_ent, q) top-11 ----
    if (tid < QT * NBEST) {
        int row = tid / NBEST, j = tid % NBEST;
        size_t o = ((size_t)s_ent * QDIM + qbase + row) * NBEST + j;
        best_sc[o] = topsc[row][j];
        best_ix[o] = topix[row][j];
    }
}

// ---------------- kernel 4: merge 4x11 candidates -> top-10 per query ----------------
__global__ __launch_bounds__(256)
void final_topk_kernel(const float* __restrict__ best_sc, const int* __restrict__ best_ix,
                       int* __restrict__ out_gid, float* __restrict__ out_sc) {
    int q = blockIdx.x * 256 + threadIdx.x;
    if (q >= QDIM) return;
    float sc[KOUT];
    int   gid[KOUT];
    #pragma unroll
    for (int k = 0; k < KOUT; ++k) { sc[k] = NEGINF; gid[k] = -1; }
    for (int se = 0; se < SDIM; ++se) {
        for (int j = 0; j < NBEST; ++j) {
            size_t o = ((size_t)se * QDIM + q) * NBEST + j;
            float s = best_sc[o];
            int  ix = best_ix[o];
            int  g;
            if (ix >= PDIM) { s += BADV; g = -1; }
            else            { g = ix * SDIM + se; }
            if (s > sc[KOUT - 1]) {
                int p = KOUT - 1;
                while (p > 0 && sc[p - 1] < s) {
                    sc[p] = sc[p - 1]; gid[p] = gid[p - 1]; --p;
                }
                sc[p] = s; gid[p] = g;
            }
        }
    }
    #pragma unroll
    for (int k = 0; k < KOUT; ++k) {
        out_gid[q * KOUT + k] = gid[k];
        out_sc [q * KOUT + k] = sc[k];
    }
}

// ---------------- host-side launcher ----------------
extern "C" void kernel_launch(void* const* d_in, const int* in_sizes, int n_in,
                              void* d_out, int out_size, void* d_ws, size_t ws_size,
                              hipStream_t stream) {
    const int*   relation = (const int*)  d_in[0];   // (S,B) int32
    const int*   head     = (const int*)  d_in[1];   // (S,B) int32
    const float* ent      = (const float*)d_in[2];   // (S,P,D) f32
    const float* rel_emb  = (const float*)d_in[3];   // (R,D) f32

    // workspace layout
    char* ws = (char*)d_ws;
    const size_t ENT_BF_BYTES  = (size_t)SDIM * PDIM * DDIM * 2;      // 64 MB
    const size_t QVEC_BYTES    = (size_t)QDIM * DDIM * 2;             // 512 KB
    const size_t BSC_BYTES     = (size_t)SDIM * QDIM * NBEST * 4;     // 176 KB
    __bf16* entbf   = (__bf16*)ws;
    __bf16* qvec    = (__bf16*)(ws + ENT_BF_BYTES);
    float*  best_sc = (float*) (ws + ENT_BF_BYTES + QVEC_BYTES);
    int*    best_ix = (int*)   (ws + ENT_BF_BYTES + QVEC_BYTES + BSC_BYTES);

    // 1) entity table fp32 -> bf16 (one HBM pass; GEMM reuse is L2-resident)
    cvt_ent_kernel<<<(SDIM * PDIM * DDIM) / 1024, 256, 0, stream>>>(ent, entbf);
    // 2) query vectors
    build_qvec_kernel<<<QDIM, 256, 0, stream>>>(relation, head, ent, rel_emb, qvec);
    // 3) fused WMMA GEMM + running top-11: 4 slices x 64 query tiles = 256 WGs
    gemm_topk_kernel<<<SDIM * (QDIM / QT), 256, 0, stream>>>(entbf, qvec, best_sc, best_ix);
    // 4) final merge; outputs: int32 gids then f32 scores, concatenated flat
    int*   out_gid = (int*)d_out;
    float* out_sc  = (float*)d_out + (size_t)SDIM * BDIM * KOUT;
    final_topk_kernel<<<(QDIM + 255) / 256, 256, 0, stream>>>(best_sc, best_ix, out_gid, out_sc);
}
